// DisableGateLSTM_37280316129948
// MI455X (gfx1250) — matrix operand order repaired and checked by reading, not compiled, not run on hardware
//
#include <hip/hip_runtime.h>
#include <hip/hip_bf16.h>
#include <math.h>

// Problem constants (from reference)
#define B_   64
#define T_   512
#define E_   256
#define H_   512
#define GD_  768          // E_ + H_ ; columns 0..511 = h, 512..767 = x
#define C_   4
#define NWG  64           // workgroups in persistent LSTM kernel
#define HSW  8            // hidden units per WG (H_/NWG)
#define ROWS 32           // 4 gates * HSW weight rows per WG

typedef __attribute__((ext_vector_type(16))) __bf16 v16bf;
typedef __attribute__((ext_vector_type(8)))  float  v8f;
typedef __attribute__((ext_vector_type(4)))  unsigned int u32x4;

__device__ __forceinline__ float sigmoidf_(float x) {
    return 1.0f / (1.0f + __expf(-x));
}
__device__ __forceinline__ float tanhf_fast(float x) {
    const float e = __expf(-2.0f * x);          // one v_exp_f32
    return (1.0f - e) / (1.0f + e);
}

// ---------------------------------------------------------------------------
// Split device-wide barrier. Exactly one barrier per scan iteration, so the
// wait target is simply gen >= t+1 (gen zeroed by init kernel each call ->
// deterministic across graph replays).
// ---------------------------------------------------------------------------
__device__ __forceinline__ void barrier_arrive(unsigned* cnt, unsigned* gen, unsigned nwg) {
    __threadfence();                 // publish this block's global stores
    __syncthreads();
    if (threadIdx.x == 0) {
        unsigned prev = __hip_atomic_fetch_add(cnt, 1u, __ATOMIC_ACQ_REL, __HIP_MEMORY_SCOPE_AGENT);
        if (prev == nwg - 1u) {
            __hip_atomic_store(cnt, 0u, __ATOMIC_RELAXED, __HIP_MEMORY_SCOPE_AGENT);
            __hip_atomic_fetch_add(gen, 1u, __ATOMIC_ACQ_REL, __HIP_MEMORY_SCOPE_AGENT);
        }
    }
}
__device__ __forceinline__ void barrier_wait(unsigned* gen, unsigned target) {
    if (threadIdx.x == 0) {
        while (__hip_atomic_load(gen, __ATOMIC_ACQUIRE, __HIP_MEMORY_SCOPE_AGENT) < target) {
            __builtin_amdgcn_s_sleep(2);
        }
    }
    __syncthreads();
}

// ---------------------------------------------------------------------------
// Kernel 0: zero h0 double-buffer + barrier counters
// ---------------------------------------------------------------------------
__global__ void lstm_init_kernel(unsigned short* hbuf, unsigned* ctrs) {
    int idx = blockIdx.x * blockDim.x + threadIdx.x;
    if (idx < 2 * B_ * H_) hbuf[idx] = 0;        // bf16 +0.0
    if (idx < 2)           ctrs[idx] = 0u;
}

// ---------------------------------------------------------------------------
// Kernel 1: embedding gather + f32 -> bf16 convert into Xbf[t][b][e]
// ---------------------------------------------------------------------------
__global__ void embed_kernel(const int* __restrict__ ids,
                             const float* __restrict__ emb,
                             unsigned short* __restrict__ Xbf_) {
    __bf16* Xbf = (__bf16*)Xbf_;
    int token = blockIdx.x;            // token = b*T_ + t  (ids is [B,T] row-major)
    int b = token / T_;
    int t = token - b * T_;
    int id = ids[token];
    int e = threadIdx.x;
    float v = emb[(size_t)id * E_ + e];
    Xbf[((size_t)t * B_ + b) * E_ + e] = (__bf16)v;
}

// ---------------------------------------------------------------------------
// One 16x16x32 bf16 WMMA K-step.
//   src     : base of this lane's A row (batch row), k-local indexing
//   klocal  : k offset within src (multiple of 32)
//   wrow    : &Wlds[(rt*16+n)*GD_]  (this lane's B output row)
//   kb      : k offset within the weight row (multiple of 32)
// A frag (16x32 bf16): lanes 0-15 K={0..7,16..23}, lanes 16-31 K={8..15,24..31}
// B frag (32x16 bf16): lanes 0-15 K={0..7,16..23}, lanes 16-31 K={8..15,24..31}
// ---------------------------------------------------------------------------
__device__ __forceinline__ v8f wmma_step(const __bf16* __restrict__ src, int klocal,
                                         const __bf16* __restrict__ wrow, int kb,
                                         int half, v8f acc) {
    union { u32x4 u[2]; v16bf v; } af;
    af.u[0] = *(const u32x4*)(src + klocal + half * 8);
    af.u[1] = *(const u32x4*)(src + klocal + half * 8 + 16);
    const __bf16* wp = wrow + kb + half * 8;
    union { u32x4 u[2]; v16bf v; } bfr;
    bfr.u[0] = *(const u32x4*)(wp);
    bfr.u[1] = *(const u32x4*)(wp + 16);
    return __builtin_amdgcn_wmma_f32_16x16x32_bf16(
        false, af.v, false, bfr.v, (short)0, acc, false, false);
}

// ---------------------------------------------------------------------------
// Kernel 2: persistent LSTM scan. 64 WGs x 256 threads (8 wave32).
// WG g owns hidden units [g*8, g*8+8) for all 4 gates.
// Per step: out[64 batch, 32 rows] = concat(h_t, x_t) @ Wslice^T via WMMA.
// The x-part (8 WMMAs) for step t+1 is computed between barrier arrive and
// barrier wait, hiding it (and the h stores) under barrier latency.
// ---------------------------------------------------------------------------
__global__ void __launch_bounds__(256)
lstm_scan_kernel(const float* __restrict__ W_f, const float* __restrict__ b_f,
                 const float* __restrict__ W_i, const float* __restrict__ b_i,
                 const float* __restrict__ W_o, const float* __restrict__ b_o,
                 const float* __restrict__ W_c, const float* __restrict__ b_c,
                 const unsigned short* __restrict__ Xbf_,
                 unsigned short* __restrict__ hbuf_,
                 float* __restrict__ hmaxbuf,
                 unsigned* ctrs) {
    __shared__ __align__(16) __bf16 Wlds[ROWS * GD_];   // 48 KB weight slice (bf16)
    __shared__ float gatesLds[ROWS * B_];               // 8 KB gate pre-activations

    const __bf16* Xbf  = (const __bf16*)Xbf_;
    __bf16*       hbuf = (__bf16*)hbuf_;

    const int tid      = threadIdx.x;
    const int hid_base = blockIdx.x * HSW;

    // ---- stage weight slice (f32 -> bf16) into LDS, once ----
    const float* Wg[4] = {W_f, W_i, W_o, W_c};
    for (int idx = tid; idx < ROWS * GD_; idx += 256) {
        int row  = idx / GD_;
        int col  = idx - row * GD_;
        int gate = row >> 3;
        int hs   = row & 7;
        Wlds[idx] = (__bf16)Wg[gate][(size_t)(hid_base + hs) * GD_ + col];
    }

    // ---- per-thread recurrent state: 2 (batch, hidden) pairs ----
    const int hs  = tid & 7;
    const int b0  = tid >> 3;          // 0..31
    const int b1  = b0 + 32;
    const int hid = hid_base + hs;
    const float bfv = b_f[hid], biv = b_i[hid], bov = b_o[hid], bcv = b_c[hid];
    float c0 = 0.f, c1 = 0.f;
    float hm0 = -1e30f, hm1 = -1e30f;

    // ---- wave/lane geometry for WMMA tiles ----
    const int lane = tid & 31;
    const int wave = tid >> 5;         // 0..7
    const int mb   = wave & 3;         // batch tile 0..3
    const int rt   = wave >> 2;        // row tile 0..1
    const int half = lane >> 4;        // 0|1
    const int n    = lane & 15;
    const int bidx = mb * 16 + (lane & 15);   // batch row handled by this lane
    const __bf16* wrow = &Wlds[(size_t)(rt * 16 + n) * GD_];

    __syncthreads();                   // Wlds ready

    // ---- pre-compute x contribution for t = 0 ----
    v8f accx = {0.f, 0.f, 0.f, 0.f, 0.f, 0.f, 0.f, 0.f};
    {
        const __bf16* xrow = Xbf + (size_t)bidx * E_;   // t = 0
        #pragma unroll
        for (int kb = H_; kb < GD_; kb += 32)
            accx = wmma_step(xrow, kb - H_, wrow, kb, half, accx);
    }

    for (int t = 0; t < T_; ++t) {
        const __bf16* hrow = hbuf + (size_t)(t & 1) * (B_ * H_) + (size_t)bidx * H_;
        __bf16*       hnxt = hbuf + (size_t)((t + 1) & 1) * (B_ * H_);

        // ---- recurrent GEMM part (16 WMMAs): k = 0..511 over h_{t-1} ----
        v8f acc = accx;
        #pragma unroll
        for (int kb = 0; kb < H_; kb += 32)
            acc = wmma_step(hrow, kb, wrow, kb, half, acc);

        // scatter D (M=batch, N=weight row) to gatesLds[row][batch]
        {
            const int mofs = half * 8;
            #pragma unroll
            for (int r = 0; r < 8; ++r) {
                gatesLds[(rt * 16 + n) * B_ + (mb * 16 + r + mofs)] = acc[r];
            }
        }
        __syncthreads();

        // ---- elementwise LSTM cell for the 2 owned (b, hs) pairs ----
        {
            const float pf0 = gatesLds[(hs)      * B_ + b0] + bfv;
            const float pi0 = gatesLds[(8 + hs)  * B_ + b0] + biv;
            const float po0 = gatesLds[(16 + hs) * B_ + b0] + bov;
            const float pg0 = gatesLds[(24 + hs) * B_ + b0] + bcv;
            const float pf1 = gatesLds[(hs)      * B_ + b1] + bfv;
            const float pi1 = gatesLds[(8 + hs)  * B_ + b1] + biv;
            const float po1 = gatesLds[(16 + hs) * B_ + b1] + bov;
            const float pg1 = gatesLds[(24 + hs) * B_ + b1] + bcv;

            c0 = sigmoidf_(pf0) * c0 + sigmoidf_(pi0) * tanhf_fast(pg0);
            c1 = sigmoidf_(pf1) * c1 + sigmoidf_(pi1) * tanhf_fast(pg1);
            const float h0 = sigmoidf_(po0) * tanhf_fast(c0);
            const float h1 = sigmoidf_(po1) * tanhf_fast(c1);
            hm0 = fmaxf(hm0, h0);
            hm1 = fmaxf(hm1, h1);
            hnxt[(size_t)b0 * H_ + hid] = (__bf16)h0;
            hnxt[(size_t)b1 * H_ + hid] = (__bf16)h1;
        }

        // ---- signal, then hide the next step's x-GEMM under barrier latency ----
        barrier_arrive(&ctrs[0], &ctrs[1], NWG);

        accx = (v8f){0.f, 0.f, 0.f, 0.f, 0.f, 0.f, 0.f, 0.f};
        if (t + 1 < T_) {
            const __bf16* xrow = Xbf + ((size_t)(t + 1) * B_ + bidx) * E_;
            if (t + 2 < T_)
                __builtin_prefetch(Xbf + ((size_t)(t + 2) * B_ + bidx) * E_, 0, 1);
            #pragma unroll
            for (int kb = H_; kb < GD_; kb += 32)
                accx = wmma_step(xrow, kb - H_, wrow, kb, half, accx);
        }

        barrier_wait(&ctrs[1], (unsigned)(t + 1));
    }

    // running max over time -> global
    hmaxbuf[(size_t)b0 * H_ + hid] = hm0;
    hmaxbuf[(size_t)b1 * H_ + hid] = hm1;
}

// ---------------------------------------------------------------------------
// Kernel 3: logits[b][c] = hmax[b,:] . fc_w[c,:] + fc_b[c]   (64x512x4, tiny)
// ---------------------------------------------------------------------------
__global__ void logits_kernel(const float* __restrict__ hmax,
                              const float* __restrict__ fc_w,
                              const float* __restrict__ fc_b,
                              float* __restrict__ out) {
    const int tid = threadIdx.x;       // 256 = 64*4
    const int b = tid >> 2;
    const int c = tid & 3;
    float s = fc_b[c];
    for (int h = 0; h < H_; ++h) {
        s += hmax[(size_t)b * H_ + h] * fc_w[(size_t)c * H_ + h];
    }
    out[b * C_ + c] = s;
}

// ---------------------------------------------------------------------------
// Host launcher
// ---------------------------------------------------------------------------
extern "C" void kernel_launch(void* const* d_in, const int* in_sizes, int n_in,
                              void* d_out, int out_size, void* d_ws, size_t ws_size,
                              hipStream_t stream) {
    (void)in_sizes; (void)n_in; (void)out_size; (void)ws_size;

    const int*   ids  = (const int*)d_in[0];
    const float* emb  = (const float*)d_in[1];
    const float* W_f  = (const float*)d_in[2];
    const float* b_f  = (const float*)d_in[3];
    const float* W_i  = (const float*)d_in[4];
    const float* b_i  = (const float*)d_in[5];
    const float* W_o  = (const float*)d_in[6];
    const float* b_o  = (const float*)d_in[7];
    const float* W_c  = (const float*)d_in[8];
    const float* b_c  = (const float*)d_in[9];
    const float* fc_w = (const float*)d_in[10];
    const float* fc_b = (const float*)d_in[11];
    float* out = (float*)d_out;

    // workspace layout
    char* ws = (char*)d_ws;
    const size_t xbf_bytes  = (size_t)T_ * B_ * E_ * 2;        // 16 MB bf16
    const size_t hbuf_bytes = (size_t)2 * B_ * H_ * 2;         // 128 KB bf16 double buffer
    const size_t hmax_bytes = (size_t)B_ * H_ * 4;             // 128 KB f32
    unsigned short* Xbf    = (unsigned short*)(ws);
    unsigned short* hbuf   = (unsigned short*)(ws + xbf_bytes);
    float*          hmax   = (float*)(ws + xbf_bytes + hbuf_bytes);
    unsigned*       ctrs   = (unsigned*)(ws + xbf_bytes + hbuf_bytes + hmax_bytes);

    // 0) deterministic init of h0 + barrier counters
    lstm_init_kernel<<<(2 * B_ * H_ + 255) / 256, 256, 0, stream>>>(hbuf, ctrs);
    // 1) embedding gather -> bf16  (one block per token)
    embed_kernel<<<B_ * T_, 256, 0, stream>>>(ids, emb, Xbf);
    // 2) persistent WMMA LSTM scan (64 WGs must be co-resident for the barrier)
    lstm_scan_kernel<<<NWG, 256, 0, stream>>>(W_f, b_f, W_i, b_i, W_o, b_o, W_c, b_c,
                                              Xbf, hbuf, hmax, ctrs);
    // 3) final classifier
    logits_kernel<<<1, 256, 0, stream>>>(hmax, fc_w, fc_b, out);
}
